// EnergyLoss_37039797960932
// MI455X (gfx1250) — matrix operand order repaired and checked by reading, not compiled, not run on hardware
//
#include <hip/hip_runtime.h>
#include <hip/hip_bf16.h>
#include <math.h>

typedef __attribute__((ext_vector_type(16))) _Float16 v16h;
typedef __attribute__((ext_vector_type(8)))  float    v8f;

// Problem constants (from reference)
#define NN   64      // transducers
#define MM   64      // focal points
#define KDIM 128     // [Ar(64) | Ai(64)]
#define COLS 512     // 8 channels per focal point * 64
#define BROWS 128    // batch rows per block
#define LDSROW 136   // 128 halfs + 8 halfs pad (16B) to dodge bank conflicts

union Frag {
    v16h v;
    float4 f4[2];
};

union H4 {
    _Float16 h[4];
    unsigned long long u;
};

// ---------------------------------------------------------------------------
// Kernel 1: build f16 weight matrix [512 channel-rows x 128 k] directly in
// WMMA *A-fragment* order: W[((ct*4 + ks)*32 + lane)*16 + idx]
//   channel row = ct*16 + (lane&15)     (row = m*8 + p)
//   A k order:  kloc = idx + ((lane&16)?16:0);  k = ks*32+kloc
//   k<64 -> coefficient of Ar[n=k] ; k>=64 -> coefficient of Ai[n=k-64]
// Scales sqrt(C1*1e5)/sqrt(-C2*1e5) folded so E*1e5 = y0^2+y1^2 - sum(y2..7^2)
// ---------------------------------------------------------------------------
__global__ void prep_weights(const double* __restrict__ h,
                             const double* __restrict__ q,
                             _Float16* __restrict__ W) {
    const double PI   = 3.14159265358979323846;
    const double Rr   = 0.000865;
    const double RHO  = 1.225;
    const double CS   = 343.0;
    const double CONSTv = 2.0 * PI * Rr * Rr * Rr;
    const double C1 = CONSTv / (6.0 * RHO * CS * CS);
    const double C2 = -CONSTv * RHO / 4.0;
    const double s1 = sqrt(C1 * 1.0e5);
    const double s2 = sqrt(-C2 * 1.0e5);

    const int total = 32 * 4 * 32 * 16;  // 65536 halfs
    for (int fi = threadIdx.x; fi < total; fi += blockDim.x) {
        int idx  = fi & 15;
        int lane = (fi >> 4) & 31;
        int ks   = (fi >> 9) & 3;
        int ct   = fi >> 11;

        int row = ct * 16 + (lane & 15);   // channel row
        int m = row >> 3;
        int p = row & 7;

        int kloc = idx + ((lane & 16) ? 16 : 0);
        int k = ks * 32 + kloc;
        int chan = k >> 6;   // 0 = multiplies Ar, 1 = multiplies Ai
        int n = k & 63;

        double re, im, s;
        if (p < 2) {
            int o = (m * NN + n) * 2;
            re = h[o]; im = h[o + 1]; s = s1;
        } else {
            int d = (p < 5) ? (p - 2) : (p - 5);
            int o = ((m * 3 + d) * NN + n) * 2;
            re = q[o]; im = q[o + 1]; s = s2;
        }
        // pi: 0 = real-part output channel, 1 = imag-part output channel
        int pi = (p == 1 || p >= 5) ? 1 : 0;
        double wv;
        if (pi == 0) wv = (chan == 0) ? ( s * re) : (-s * im);
        else         wv = (chan == 0) ? ( s * im) : ( s * re);

        W[fi] = (_Float16)wv;
    }
}

// ---------------------------------------------------------------------------
// One 16x16 tile: D[channel, batch] = sum_ks A(weights)[ks] x B(X^T)[ks]
// Returns per-lane E*1e5 = c0^2 + c1^2 - (c2^2 + ... + c7^2)
// ---------------------------------------------------------------------------
__device__ __forceinline__ float tile_energy(const v16h a0, const v16h a1,
                                             const v16h a2, const v16h a3,
                                             const v16h b0, const v16h b1,
                                             const v16h b2, const v16h b3) {
    v8f c = {};
    c = __builtin_amdgcn_wmma_f32_16x16x32_f16(false, a0, false, b0,
                                               (short)0, c, false, false);
    c = __builtin_amdgcn_wmma_f32_16x16x32_f16(false, a1, false, b1,
                                               (short)0, c, false, false);
    c = __builtin_amdgcn_wmma_f32_16x16x32_f16(false, a2, false, b2,
                                               (short)0, c, false, false);
    c = __builtin_amdgcn_wmma_f32_16x16x32_f16(false, a3, false, b3,
                                               (short)0, c, false, false);
    float pos = fmaf(c[0], c[0], c[1] * c[1]);
    float neg = c[2] * c[2];
    neg = fmaf(c[3], c[3], neg);
    neg = fmaf(c[4], c[4], neg);
    neg = fmaf(c[5], c[5], neg);
    neg = fmaf(c[6], c[6], neg);
    neg = fmaf(c[7], c[7], neg);
    return pos - neg;
}

// ---------------------------------------------------------------------------
// Kernel 2: main GEMM + energy + loss partials.
// Block: 256 threads (8 waves), 128 batch rows. Grid: B/128 blocks.
// Wave: owns 16 batch rows (B-matrix, resident in regs); loops 32 channel
// tiles (A-matrix streamed, double-buffered). All 32 lanes produce one
// (b,m) energy per tile -> no cross-lane ops in the hot loop.
// ---------------------------------------------------------------------------
__global__ __launch_bounds__(256) void energy_main(
        const float* __restrict__ amp,
        const float* __restrict__ ph,
        const float* __restrict__ trueE,
        const _Float16* __restrict__ W,
        float* __restrict__ partial) {
    __shared__ _Float16 Xh[BROWS * LDSROW];
    __shared__ float wsum[8];

    const int t  = threadIdx.x;
    const int b0 = blockIdx.x * BROWS;

    // ---- Stage 1: activations X[b,128] = [amp*cos(phi) | amp*sin(phi)] -> LDS
    // Vectorized: b128 global loads (the dominant HBM streams), b64 LDS stores.
    {
        const float4* amp4 = reinterpret_cast<const float4*>(amp + (size_t)b0 * NN);
        const float4* ph4  = reinterpret_cast<const float4*>(ph  + (size_t)b0 * NN);
        #pragma unroll
        for (int i = 0; i < 8; ++i) {
            int fi4 = i * 256 + t;           // 2048 float4s per block, coalesced
            int bl  = fi4 >> 4;              // 16 float4 per batch row
            int n   = (fi4 & 15) * 4;
            float4 a4 = amp4[fi4];
            float4 p4 = ph4[fi4];
            H4 cr, si;
            float s, c;
            __sincosf(p4.x, &s, &c); cr.h[0] = (_Float16)(a4.x * c); si.h[0] = (_Float16)(a4.x * s);
            __sincosf(p4.y, &s, &c); cr.h[1] = (_Float16)(a4.y * c); si.h[1] = (_Float16)(a4.y * s);
            __sincosf(p4.z, &s, &c); cr.h[2] = (_Float16)(a4.z * c); si.h[2] = (_Float16)(a4.z * s);
            __sincosf(p4.w, &s, &c); cr.h[3] = (_Float16)(a4.w * c); si.h[3] = (_Float16)(a4.w * s);
            *reinterpret_cast<unsigned long long*>(&Xh[bl * LDSROW + n])      = cr.u;
            *reinterpret_cast<unsigned long long*>(&Xh[bl * LDSROW + 64 + n]) = si.u;
        }
    }
    __syncthreads();

    const int lane = t & 31;
    const int w    = t >> 5;
    const int bl   = w * 16 + (lane & 15);   // this lane's batch column
    const int kh   = (lane & 16) ? 16 : 0;   // B-fragment K half per lane group

    // ---- Stage 2: B fragments (X^T, 32x16 per K-step) from LDS, kept in regs.
    // B layout: lane<16 -> col=lane, K=idx; lane>=16 -> col=lane-16, K=16+idx.
    // => 32 contiguous bytes of Xh row `bl` starting at k = ks*32 + kh.
    Frag bfr[4];
    #pragma unroll
    for (int ks = 0; ks < 4; ++ks) {
        const _Float16* base = &Xh[bl * LDSROW + ks * 32 + kh];
        bfr[ks].f4[0] = *reinterpret_cast<const float4*>(base);
        bfr[ks].f4[1] = *reinterpret_cast<const float4*>(base + 8);
    }

    const v16h* Wv = reinterpret_cast<const v16h*>(W);
    const int   bg = b0 + bl;                // global batch row
    const int   mo = (lane >> 4) & 1;        // m offset within tile (0 or 1)
    float acc = 0.0f;

    // ---- Stage 3: 32 channel tiles, double-buffered A-fragment stream.
    v16h aA0 = Wv[0 * 32 + lane];
    v16h aA1 = Wv[1 * 32 + lane];
    v16h aA2 = Wv[2 * 32 + lane];
    v16h aA3 = Wv[3 * 32 + lane];

    #pragma unroll 1
    for (int ct = 0; ct < 32; ct += 2) {
        // prefetch tile ct+1
        v16h aB0 = Wv[((ct + 1) * 4 + 0) * 32 + lane];
        v16h aB1 = Wv[((ct + 1) * 4 + 1) * 32 + lane];
        v16h aB2 = Wv[((ct + 1) * 4 + 2) * 32 + lane];
        v16h aB3 = Wv[((ct + 1) * 4 + 3) * 32 + lane];
        float tv0 = trueE[bg * MM + 2 * ct + mo];

        float e0 = tile_energy(aA0, aA1, aA2, aA3,
                               bfr[0].v, bfr[1].v, bfr[2].v, bfr[3].v);
        float d0 = e0 - tv0;
        acc = fmaf(d0, d0, acc);

        // prefetch tile ct+2
        if (ct + 2 < 32) {
            aA0 = Wv[((ct + 2) * 4 + 0) * 32 + lane];
            aA1 = Wv[((ct + 2) * 4 + 1) * 32 + lane];
            aA2 = Wv[((ct + 2) * 4 + 2) * 32 + lane];
            aA3 = Wv[((ct + 2) * 4 + 3) * 32 + lane];
        }
        float tv1 = trueE[bg * MM + 2 * (ct + 1) + mo];

        float e1 = tile_energy(aB0, aB1, aB2, aB3,
                               bfr[0].v, bfr[1].v, bfr[2].v, bfr[3].v);
        float d1 = e1 - tv1;
        acc = fmaf(d1, d1, acc);
    }

    // ---- Stage 4: reduce loss partials (only at the very end)
    #pragma unroll
    for (int off = 16; off > 0; off >>= 1) acc += __shfl_xor(acc, off, 32);
    if (lane == 0) wsum[w] = acc;
    __syncthreads();
    if (t == 0) {
        float s = 0.0f;
        #pragma unroll
        for (int i = 0; i < 8; ++i) s += wsum[i];
        partial[blockIdx.x] = s;
    }
}

// ---------------------------------------------------------------------------
// Kernel 3: deterministic final reduction -> mean
// ---------------------------------------------------------------------------
__global__ void finalize_loss(const float* __restrict__ partial, int nblk,
                              float* __restrict__ out, float invcount) {
    __shared__ float red[256];
    float s = 0.0f;
    for (int i = threadIdx.x; i < nblk; i += 256) s += partial[i];
    red[threadIdx.x] = s;
    __syncthreads();
    for (int o = 128; o > 0; o >>= 1) {
        if (threadIdx.x < o) red[threadIdx.x] += red[threadIdx.x + o];
        __syncthreads();
    }
    if (threadIdx.x == 0) out[0] = red[0] * invcount;
}

extern "C" void kernel_launch(void* const* d_in, const int* in_sizes, int n_in,
                              void* d_out, int out_size, void* d_ws, size_t ws_size,
                              hipStream_t stream) {
    const float*  amp = (const float*)d_in[0];
    const float*  ph  = (const float*)d_in[1];
    const float*  te  = (const float*)d_in[2];
    const double* h   = (const double*)d_in[3];   // complex128 [64,64]
    const double* q   = (const double*)d_in[4];   // complex128 [64,3,64]

    const int B    = in_sizes[0] / NN;            // 32768
    const int nblk = B / BROWS;                   // 256

    _Float16* W     = (_Float16*)d_ws;                               // 128 KB
    float* partials = (float*)((char*)d_ws + (size_t)COLS * KDIM * 2);

    prep_weights<<<1, 256, 0, stream>>>(h, q, W);
    energy_main<<<nblk, 256, 0, stream>>>(amp, ph, te, W, partials);
    finalize_loss<<<1, 256, 0, stream>>>(partials, nblk, (float*)d_out,
                                         1.0f / (float)(B * MM));
}
